// SpikeTDANetLayer_30236569764340
// MI455X (gfx1250) — matrix operand order repaired and checked by reading, not compiled, not run on hardware
//
#include <hip/hip_runtime.h>
#include <math.h>

// ---------------- problem constants (from reference) ----------------
#define T_  8
#define N_  10000
#define C_  128
#define H_  4
#define WW_ 4
#define KD_ 5
#define E_  80000
#define DH_ 32
#define M_  (T_ * N_)     // 80000 rows for all row-wise GEMMs
#define C4_ (4 * C_)      // 512

#define THETA_ 1.0f
#define GAMMA_ 5.0f
#define BETA_  0.9f
#define EPS_   1e-5f

// ---------------- types ----------------
typedef __bf16 bf16;
typedef bf16  bf16x8 __attribute__((ext_vector_type(8)));
typedef bf16  v16bf  __attribute__((ext_vector_type(16)));
typedef float v8f    __attribute__((ext_vector_type(8)));
typedef int   int4v  __attribute__((ext_vector_type(4)));

#if __has_builtin(__builtin_amdgcn_global_load_async_to_lds_b128)
#define HAVE_ASYNC_LDS 1
#define AS1_ __attribute__((address_space(1)))
#define AS3_ __attribute__((address_space(3)))
#endif

__device__ __forceinline__ bf16 tobf(float f) {
  unsigned u = __builtin_bit_cast(unsigned, f);
  unsigned r = u + 0x7FFFu + ((u >> 16) & 1u);      // round-to-nearest-even
  unsigned short h = (unsigned short)(r >> 16);
  return __builtin_bit_cast(bf16, h);
}

__device__ __forceinline__ float wave_allreduce_sum(float s) {
#pragma unroll
  for (int m = 16; m > 0; m >>= 1) s += __shfl_xor(s, m, 32);
  return s;
}

// monotonic float<->uint mapping for atomicMax on floats
__device__ __forceinline__ unsigned fmap(float f) {
  unsigned u = __builtin_bit_cast(unsigned, f);
  unsigned mask = (unsigned)(((int)u >> 31)) | 0x80000000u;
  return u ^ mask;
}
__device__ __forceinline__ float funmap(unsigned m) {
  unsigned u = (m & 0x80000000u) ? (m ^ 0x80000000u) : ~m;
  return __builtin_bit_cast(float, u);
}

__device__ __forceinline__ float gelu_exact(float x) {
  return 0.5f * x * (1.0f + erff(x * 0.70710678118654752f));
}

// ---------------- utility kernels ----------------
__global__ __launch_bounds__(256) void zero_f32_kernel(float* __restrict__ p, long n) {
  long i = (long)blockIdx.x * 256 + threadIdx.x;
  if (i < n) p[i] = 0.0f;
}

__global__ __launch_bounds__(256) void fill_negmax_kernel(unsigned* __restrict__ p, long n) {
  long i = (long)blockIdx.x * 256 + threadIdx.x;
  if (i < n) p[i] = fmap(-1.0e9f);
}

// convert weight W[K,N] (f32 row-major) -> Wt[N,K] bf16 (transposed)
__global__ __launch_bounds__(256) void cvt_w_t_kernel(const float* __restrict__ W,
                                                      bf16* __restrict__ Wt,
                                                      int Kd, int Nc) {
  int i = blockIdx.x * 256 + threadIdx.x;
  if (i < Kd * Nc) {
    int k = i / Nc, n = i % Nc;
    Wt[(size_t)n * Kd + k] = tobf(W[i]);
  }
}

// plain f32 -> bf16
__global__ __launch_bounds__(256) void cvt_f2b_kernel(const float* __restrict__ s,
                                                      bf16* __restrict__ d, long n) {
  long i = (long)blockIdx.x * 256 + threadIdx.x;
  if (i < n) d[i] = tobf(s[i]);
}

// ---------------- graph aggregation ----------------
__global__ __launch_bounds__(256) void deg_kernel(const int* __restrict__ dst,
                                                  float* __restrict__ deg) {
  int e = blockIdx.x * 256 + threadIdx.x;
  if (e < E_) atomicAdd(&deg[dst[e]], 1.0f);
}

// agg[t, dst[e], :] += x[t, src[e], :]   (thread = (t,e,c4) -> 4 floats)
__global__ __launch_bounds__(256) void scatter_agg_kernel(const float* __restrict__ x,
                                                          const int* __restrict__ src,
                                                          const int* __restrict__ dst,
                                                          float* __restrict__ agg) {
  long i = (long)blockIdx.x * 256 + threadIdx.x;   // T_*E_*32 threads exactly
  int c4 = (int)(i & 31);
  long rest = i >> 5;
  int e = (int)(rest % E_);
  int t = (int)(rest / E_);
  int s = src[e], d = dst[e];
  const float4 xv = *(const float4*)&x[((size_t)t * N_ + s) * C_ + c4 * 4];
  float* ap = &agg[((size_t)t * N_ + d) * C_ + c4 * 4];
  atomicAdd(ap + 0, xv.x);
  atomicAdd(ap + 1, xv.y);
  atomicAdd(ap + 2, xv.z);
  atomicAdd(ap + 3, xv.w);
}

// abf[i] = bf16(agg[i] / max(deg,1))
__global__ __launch_bounds__(256) void cvt_agg_kernel(const float* __restrict__ agg,
                                                      const float* __restrict__ deg,
                                                      bf16* __restrict__ abf) {
  long i = (long)blockIdx.x * 256 + threadIdx.x;   // M_*C_
  if (i < (long)M_ * C_) {
    int node = (int)((i / C_) % N_);
    abf[i] = tobf(agg[i] / fmaxf(deg[node], 1.0f));
  }
}

// ---------------- WMMA bf16 GEMM ----------------
// D[M,Ncol] = act(A[M,K] @ B[K,Ncol] + bias);  A row-major bf16, Bt = B^T row-major bf16.
// Block: 256 thr = 8 waves, tile 128(M) x 128(N), BK=32.  Wave w: wm=w>>1 owns a
// 32(M)x64(N) strip -> 2 A-frags x 4 B-frags -> 8 v_wmma_f32_16x16x32_bf16 per K-step,
// all fragments loaded before the WMMA burst.  f32 accumulation (v8f x 8).
template <int KDIM, bool GELU, bool STOREBF>
__global__ __launch_bounds__(256) void gemm_wmma_bf16(const bf16* __restrict__ A,
                                                      const bf16* __restrict__ Bt,
                                                      const float* __restrict__ bias,
                                                      float* __restrict__ Df,
                                                      bf16* __restrict__ Dh,
                                                      int Ncol) {
  __shared__ bf16 As[128][40];   // padded: 80B row stride, 16B aligned
  __shared__ bf16 Bs[128][40];   // n-major (transposed) tile

  const int tid = threadIdx.x;
  const int w = tid >> 5, l = tid & 31;
  const int wm = w >> 1, wn = w & 1;
  const int half = l >> 4, lr = l & 15;
  const int bm0 = blockIdx.y * 128;
  const int bn0 = blockIdx.x * 128;

  // ISA 7.12.2 16-bit A layout: lanes 0-15 K in {0..7,16..23}; lanes 16-31 K in {8..15,24..31}
  const int aC0 = half * 8;
  const int aC1 = 16 + half * 8;
  // B (n-major): same K layout as A: lanes 0-15 K in {0..7,16..23}; lanes 16-31 K in {8..15,24..31}
  const int bC0 = half * 8;
  const int ar0 = wm * 32 + lr, ar1 = ar0 + 16;

  union F16 { v16bf v; bf16x8 h[2]; };

  v8f acc[2][4] = {};

  // tile loaders: 128 rows x 32 k, 2 threads per row, 16 bf16 each (two b128)
  const int ld_r = tid >> 1, ld_c = (tid & 1) * 16;
  const bf16* gA = &A[(size_t)(bm0 + ld_r) * KDIM + ld_c];
  const bf16* gB = &Bt[(size_t)(bn0 + ld_r) * KDIM + ld_c];

#pragma unroll
  for (int kk = 0; kk < KDIM; kk += 32) {
    __syncthreads();
#ifdef HAVE_ASYNC_LDS
    // CDNA5 async tile staging: GLOBAL_LOAD_ASYNC_TO_LDS_B128 (ASYNCcnt), no VGPR bounce
    __builtin_amdgcn_global_load_async_to_lds_b128(
        (AS1_ int4v*)(gA + kk),     (AS3_ int4v*)&As[ld_r][ld_c],     0, 0);
    __builtin_amdgcn_global_load_async_to_lds_b128(
        (AS1_ int4v*)(gA + kk + 8), (AS3_ int4v*)&As[ld_r][ld_c + 8], 0, 0);
    __builtin_amdgcn_global_load_async_to_lds_b128(
        (AS1_ int4v*)(gB + kk),     (AS3_ int4v*)&Bs[ld_r][ld_c],     0, 0);
    __builtin_amdgcn_global_load_async_to_lds_b128(
        (AS1_ int4v*)(gB + kk + 8), (AS3_ int4v*)&Bs[ld_r][ld_c + 8], 0, 0);
#if __has_builtin(__builtin_amdgcn_s_wait_asynccnt)
    __builtin_amdgcn_s_wait_asynccnt(0);
#else
    asm volatile("s_wait_asynccnt 0" ::: "memory");
#endif
#else
    *(bf16x8*)&As[ld_r][ld_c]     = *(const bf16x8*)(gA + kk);
    *(bf16x8*)&As[ld_r][ld_c + 8] = *(const bf16x8*)(gA + kk + 8);
    *(bf16x8*)&Bs[ld_r][ld_c]     = *(const bf16x8*)(gB + kk);
    *(bf16x8*)&Bs[ld_r][ld_c + 8] = *(const bf16x8*)(gB + kk + 8);
#endif
    __syncthreads();

    // load ALL fragments first, then burst 8 WMMAs back-to-back
    F16 af[2], bfr[4];
    af[0].h[0] = *(const bf16x8*)&As[ar0][aC0];
    af[0].h[1] = *(const bf16x8*)&As[ar0][aC1];
    af[1].h[0] = *(const bf16x8*)&As[ar1][aC0];
    af[1].h[1] = *(const bf16x8*)&As[ar1][aC1];
#pragma unroll
    for (int s = 0; s < 4; ++s) {
      const int br = wn * 64 + s * 16 + lr;
      bfr[s].h[0] = *(const bf16x8*)&Bs[br][bC0];
      bfr[s].h[1] = *(const bf16x8*)&Bs[br][bC0 + 16];
    }
#pragma unroll
    for (int s = 0; s < 4; ++s) {
#pragma unroll
      for (int m2 = 0; m2 < 2; ++m2) {
        acc[m2][s] = __builtin_amdgcn_wmma_f32_16x16x32_bf16(
            false, af[m2].v, false, bfr[s].v, (short)0, acc[m2][s], false, false);
      }
    }
  }

  // C/D layout: lane l reg r -> m = r + 8*(l>>4), n = l&15
#pragma unroll
  for (int s = 0; s < 4; ++s) {
    const int col = bn0 + wn * 64 + s * 16 + lr;
    const float bv = bias[col];
#pragma unroll
    for (int m2 = 0; m2 < 2; ++m2) {
#pragma unroll
      for (int r = 0; r < 8; ++r) {
        const int row = bm0 + wm * 32 + m2 * 16 + half * 8 + r;
        float val = acc[m2][s][r] + bv;
        if (GELU) val = gelu_exact(val);
        const size_t o = (size_t)row * Ncol + col;
        if (STOREBF) Dh[o] = tobf(val);
        else         Df[o] = val;
      }
    }
  }
}

// ---------------- LayerNorm (wave32 per row of 128) ----------------
__global__ __launch_bounds__(256) void add_ln_kernel(const float* __restrict__ a,
                                                     const float* __restrict__ b,
                                                     const float* __restrict__ g,
                                                     const float* __restrict__ bet,
                                                     float* __restrict__ out) {
  const int w = threadIdx.x >> 5, l = threadIdx.x & 31;
  const long row = (long)blockIdx.x * 8 + w;     // rows = M_ (exact multiple of 8)
  const float* ar = a + row * C_;
  const float* br = b + row * C_;
  float xv[4];
#pragma unroll
  for (int j = 0; j < 4; ++j) xv[j] = ar[l + 32 * j] + br[l + 32 * j];
  float mu = wave_allreduce_sum(xv[0] + xv[1] + xv[2] + xv[3]) * (1.0f / C_);
  float vs = 0.0f;
#pragma unroll
  for (int j = 0; j < 4; ++j) { float d = xv[j] - mu; vs += d * d; }
  float var = wave_allreduce_sum(vs) * (1.0f / C_);
  float rinv = rsqrtf(var + EPS_);
  float* orow = out + row * C_;
#pragma unroll
  for (int j = 0; j < 4; ++j) {
    int c = l + 32 * j;
    orow[c] = (xv[j] - mu) * rinv * g[c] + bet[c];
  }
}

// h2 = LN(h1 + sum_k delay_k[k]*h1[t-k]); write bf16 (feeds Q/K/V GEMMs)
__global__ __launch_bounds__(256) void delay_ln_kernel(const float* __restrict__ h1,
                                                       const float* __restrict__ dk,
                                                       const float* __restrict__ g,
                                                       const float* __restrict__ bet,
                                                       bf16* __restrict__ outb) {
  const int w = threadIdx.x >> 5, l = threadIdx.x & 31;
  const long row = (long)blockIdx.x * 8 + w;
  const int t = (int)(row / N_);
  float xv[4];
#pragma unroll
  for (int j = 0; j < 4; ++j) {
    int c = l + 32 * j;
    float v = h1[row * C_ + c];
    float xd = 0.0f;
#pragma unroll
    for (int k = 0; k < KD_; ++k) {
      if (t - k >= 0) xd += dk[k * C_ + c] * h1[(row - (long)k * N_) * C_ + c];
    }
    xv[j] = v + xd;
  }
  float mu = wave_allreduce_sum(xv[0] + xv[1] + xv[2] + xv[3]) * (1.0f / C_);
  float vs = 0.0f;
#pragma unroll
  for (int j = 0; j < 4; ++j) { float d = xv[j] - mu; vs += d * d; }
  float var = wave_allreduce_sum(vs) * (1.0f / C_);
  float rinv = rsqrtf(var + EPS_);
#pragma unroll
  for (int j = 0; j < 4; ++j) {
    int c = l + 32 * j;
    outb[row * C_ + c] = tobf((xv[j] - mu) * rinv * g[c] + bet[c]);
  }
}

// ---------------- edge attention ----------------
// s_all[t][e][w][h]
__global__ __launch_bounds__(256) void edge_scores_kernel(const float* __restrict__ q,
                                                          const float* __restrict__ k,
                                                          const int* __restrict__ src,
                                                          const int* __restrict__ dst,
                                                          float* __restrict__ sall) {
  long idx = (long)blockIdx.x * 256 + threadIdx.x;   // T_*E_*WW_*H_ exact
  int h = (int)(idx & 3);
  int ww = (int)((idx >> 2) & 3);
  long rest = idx >> 4;
  int e = (int)(rest % E_);
  int t = (int)(rest / E_);
  float s;
  if (t < ww) {
    s = -1.0e9f;
  } else {
    int sd = src[e], dd = dst[e];
    const float4* q4 = (const float4*)&q[(((size_t)t * N_ + dd) * H_ + h) * DH_];
    const float4* k4 = (const float4*)&k[(((size_t)(t - ww) * N_ + sd) * H_ + h) * DH_];
    float acc = 0.0f;
#pragma unroll
    for (int i = 0; i < 8; ++i) {
      float4 a = q4[i], b = k4[i];
      acc += a.x * b.x + a.y * b.y + a.z * b.z + a.w * b.w;
    }
    s = acc * 0.17677669529663687f;   // 1/sqrt(32)
  }
  sall[idx] = s;
}

__global__ __launch_bounds__(256) void segmax_kernel(const float* __restrict__ sall,
                                                     const int* __restrict__ dst,
                                                     unsigned* __restrict__ mbuf) {
  long idx = (long)blockIdx.x * 256 + threadIdx.x;   // T_*E_*H_ exact
  int h = (int)(idx & 3);
  long rest = idx >> 2;
  int e = (int)(rest % E_);
  int t = (int)(rest / E_);
  const float* sp = &sall[(((size_t)t * E_ + e) * WW_) * H_ + h];
  float mw = sp[0];
#pragma unroll
  for (int w = 1; w < WW_; ++w) mw = fmaxf(mw, sp[w * H_]);
  atomicMax(&mbuf[((size_t)t * N_ + dst[e]) * H_ + h], fmap(mw));
}

__global__ __launch_bounds__(256) void expden_kernel(float* __restrict__ sall,
                                                     const int* __restrict__ dst,
                                                     const unsigned* __restrict__ mbuf,
                                                     float* __restrict__ den) {
  long idx = (long)blockIdx.x * 256 + threadIdx.x;   // T_*E_*WW_*H_ exact
  int h = (int)(idx & 3);
  long rest = idx >> 4;
  int e = (int)(rest % E_);
  int t = (int)(rest / E_);
  int dd = dst[e];
  float m = funmap(mbuf[((size_t)t * N_ + dd) * H_ + h]);
  float ev = expf(sall[idx] - m);
  sall[idx] = ev;
  atomicAdd(&den[((size_t)t * N_ + dd) * H_ + h], ev);
}

__global__ __launch_bounds__(256) void attn_out_kernel(const float* __restrict__ sall,
                                                       const float* __restrict__ den,
                                                       const float* __restrict__ spikes,
                                                       const float* __restrict__ v,
                                                       const int* __restrict__ src,
                                                       const int* __restrict__ dst,
                                                       float* __restrict__ aout) {
  long idx = (long)blockIdx.x * 256 + threadIdx.x;   // T_*E_*WW_*H_ exact
  int h = (int)(idx & 3);
  int ww = (int)((idx >> 2) & 3);
  long rest = idx >> 4;
  int e = (int)(rest % E_);
  int t = (int)(rest / E_);
  if (t < ww) return;                                 // shifted spike gate is 0
  int sd = src[e], dd = dst[e];
  float gate = spikes[(size_t)(t - ww) * N_ + sd];
  float wgt = sall[idx] / (den[((size_t)t * N_ + dd) * H_ + h] + 1e-9f);
  float cf = wgt * gate;
  if (cf == 0.0f) return;
  const float4* v4 = (const float4*)&v[(((size_t)(t - ww) * N_ + sd) * H_ + h) * DH_];
  float* op = &aout[(((size_t)t * N_ + dd) * H_ + h) * DH_];
#pragma unroll
  for (int i = 0; i < 8; ++i) {
    float4 vv = v4[i];
    atomicAdd(op + i * 4 + 0, cf * vv.x);
    atomicAdd(op + i * 4 + 1, cf * vv.y);
    atomicAdd(op + i * 4 + 2, cf * vv.z);
    atomicAdd(op + i * 4 + 3, cf * vv.w);
  }
}

// ---------------- LIF head ----------------
__global__ __launch_bounds__(256) void imsg_kernel(const float* __restrict__ msg,
                                                   const float* __restrict__ wmsg,
                                                   const float* __restrict__ bmsg,
                                                   float* __restrict__ Ibuf, int rows) {
  const int w = threadIdx.x >> 5, l = threadIdx.x & 31;
  const long row = (long)blockIdx.x * 8 + w;
  if (row >= rows) return;
  const float* mr = msg + row * C_;
  float s = 0.0f;
#pragma unroll
  for (int j = 0; j < 4; ++j) { int c = l + 32 * j; s += mr[c] * wmsg[c]; }
  s = wave_allreduce_sum(s);
  if (l == 0) Ibuf[row] = s + bmsg[0];
}

__global__ __launch_bounds__(256) void lif_kernel(const float* __restrict__ Ibuf,
                                                  float* __restrict__ spk_out) {
  int n = blockIdx.x * 256 + threadIdx.x;
  if (n >= N_) return;
  float vv = 0.0f, ss = 0.0f;
#pragma unroll
  for (int t = 0; t < T_; ++t) {
    float It = Ibuf[(size_t)t * N_ + n];
    float vm = BETA_ * vv * (1.0f - ss) + It;
    float sur = 1.0f / (1.0f + expf(-GAMMA_ * (vm - THETA_)));
    float hard = (vm > THETA_) ? 1.0f : 0.0f;
    float spk = (hard - sur) + sur;     // matches stop-gradient trick arithmetic
    spk_out[(size_t)t * N_ + n] = spk;
    vv = vm; ss = spk;
  }
}

// ---------------- host launch ----------------
static inline unsigned nblk(long n, int b) { return (unsigned)((n + b - 1) / b); }

extern "C" void kernel_launch(void* const* d_in, const int* in_sizes, int n_in,
                              void* d_out, int out_size, void* d_ws, size_t ws_size,
                              hipStream_t stream) {
  (void)in_sizes; (void)n_in; (void)out_size; (void)ws_size;

  const float* x      = (const float*)d_in[0];
  const float* spikes = (const float*)d_in[1];
  const int*   ei     = (const int*)d_in[2];
  const int*   srcp   = ei;
  const int*   dstp   = ei + E_;
  const float* W_gnn  = (const float*)d_in[4];
  const float* b_gnn  = (const float*)d_in[5];
  const float* g1     = (const float*)d_in[6];
  const float* b1     = (const float*)d_in[7];
  const float* delayk = (const float*)d_in[8];
  const float* g2     = (const float*)d_in[9];
  const float* b2     = (const float*)d_in[10];
  const float* Wq = (const float*)d_in[11]; const float* bq = (const float*)d_in[12];
  const float* Wk = (const float*)d_in[13]; const float* bk = (const float*)d_in[14];
  const float* Wv = (const float*)d_in[15]; const float* bv = (const float*)d_in[16];
  const float* Wo = (const float*)d_in[17]; const float* bo = (const float*)d_in[18];
  const float* Wmsg = (const float*)d_in[19]; const float* bmsg = (const float*)d_in[20];
  const float* Wfc1 = (const float*)d_in[21]; const float* bfc1 = (const float*)d_in[22];
  const float* Wfc2 = (const float*)d_in[23]; const float* bfc2 = (const float*)d_in[24];
  const float* gf = (const float*)d_in[25];  const float* bff = (const float*)d_in[26];

  float* feat    = (float*)d_out;                 // (T,N,C)
  float* spk_out = feat + (size_t)M_ * C_;        // (T,N)

  // ---- carve workspace ----
  char* base = (char*)d_ws;
  size_t off = 0;
  auto carve = [&](size_t bytes) -> void* {
    off = (off + 255) & ~(size_t)255;
    void* p = base + off;
    off += bytes;
    return p;
  };
  float*    deg   = (float*)carve((size_t)N_ * 4);
  float*    agg   = (float*)carve((size_t)M_ * C_ * 4);
  bf16*     abf   = (bf16*)carve((size_t)M_ * C4_ * 2);   // staging A (bf16)
  bf16*     bbf   = (bf16*)carve((size_t)M_ * C4_ * 2);   // fc1 gelu output (bf16)
  bf16*     wgnnT = (bf16*)carve((size_t)C_ * C_ * 2);
  bf16*     wqT   = (bf16*)carve((size_t)C_ * C_ * 2);
  bf16*     wkT   = (bf16*)carve((size_t)C_ * C_ * 2);
  bf16*     wvT   = (bf16*)carve((size_t)C_ * C_ * 2);
  bf16*     woT   = (bf16*)carve((size_t)C_ * C_ * 2);
  bf16*     wfc1T = (bf16*)carve((size_t)C_ * C4_ * 2);
  bf16*     wfc2T = (bf16*)carve((size_t)C4_ * C_ * 2);
  float*    xsp   = (float*)carve((size_t)M_ * C_ * 4);
  float*    h1    = (float*)carve((size_t)M_ * C_ * 4);
  float*    qb    = (float*)carve((size_t)M_ * C_ * 4);
  float*    kb    = (float*)carve((size_t)M_ * C_ * 4);
  float*    vb    = (float*)carve((size_t)M_ * C_ * 4);
  float*    sall  = (float*)carve((size_t)T_ * E_ * WW_ * H_ * 4);
  unsigned* mbuf  = (unsigned*)carve((size_t)T_ * N_ * H_ * 4);
  float*    den   = (float*)carve((size_t)T_ * N_ * H_ * 4);
  float*    aout  = (float*)carve((size_t)M_ * C_ * 4);
  float*    msg   = (float*)carve((size_t)M_ * C_ * 4);
  float*    Ibuf  = (float*)carve((size_t)T_ * N_ * 4);
  float*    ffn   = (float*)carve((size_t)M_ * C_ * 4);

  const long MC   = (long)M_ * C_;
  const long TEWH = (long)T_ * E_ * WW_ * H_;
  const long TEH  = (long)T_ * E_ * H_;
  const long TNH  = (long)T_ * N_ * H_;

  // ---- init accumulators (must happen every call) ----
  zero_f32_kernel<<<nblk(N_, 256), 256, 0, stream>>>(deg, N_);
  zero_f32_kernel<<<nblk(MC, 256), 256, 0, stream>>>(agg, MC);
  zero_f32_kernel<<<nblk(TNH, 256), 256, 0, stream>>>(den, TNH);
  zero_f32_kernel<<<nblk(MC, 256), 256, 0, stream>>>(aout, MC);
  fill_negmax_kernel<<<nblk(TNH, 256), 256, 0, stream>>>(mbuf, TNH);

  // ---- weight conversion (bf16, transposed) ----
  cvt_w_t_kernel<<<nblk(C_ * C_, 256), 256, 0, stream>>>(W_gnn, wgnnT, C_, C_);
  cvt_w_t_kernel<<<nblk(C_ * C_, 256), 256, 0, stream>>>(Wq, wqT, C_, C_);
  cvt_w_t_kernel<<<nblk(C_ * C_, 256), 256, 0, stream>>>(Wk, wkT, C_, C_);
  cvt_w_t_kernel<<<nblk(C_ * C_, 256), 256, 0, stream>>>(Wv, wvT, C_, C_);
  cvt_w_t_kernel<<<nblk(C_ * C_, 256), 256, 0, stream>>>(Wo, woT, C_, C_);
  cvt_w_t_kernel<<<nblk(C_ * C4_, 256), 256, 0, stream>>>(Wfc1, wfc1T, C_, C4_);
  cvt_w_t_kernel<<<nblk(C4_ * C_, 256), 256, 0, stream>>>(Wfc2, wfc2T, C4_, C_);

  // ---- GNN aggregation ----
  deg_kernel<<<nblk(E_, 256), 256, 0, stream>>>(dstp, deg);
  scatter_agg_kernel<<<nblk((long)T_ * E_ * 32, 256), 256, 0, stream>>>(x, srcp, dstp, agg);
  cvt_agg_kernel<<<nblk(MC, 256), 256, 0, stream>>>(agg, deg, abf);

  // x_spatial = (agg/deg) @ W_gnn + b_gnn
  gemm_wmma_bf16<128, false, false><<<dim3(1, M_ / 128), 256, 0, stream>>>(
      abf, wgnnT, b_gnn, xsp, nullptr, C_);

  // h1 = LN(x + x_spatial)
  add_ln_kernel<<<M_ / 8, 256, 0, stream>>>(x, xsp, g1, b1, h1);

  // h2 = LN(h1 + delay conv) -> bf16 staging
  delay_ln_kernel<<<M_ / 8, 256, 0, stream>>>(h1, delayk, g2, b2, abf);

  // Q/K/V
  gemm_wmma_bf16<128, false, false><<<dim3(1, M_ / 128), 256, 0, stream>>>(
      abf, wqT, bq, qb, nullptr, C_);
  gemm_wmma_bf16<128, false, false><<<dim3(1, M_ / 128), 256, 0, stream>>>(
      abf, wkT, bk, kb, nullptr, C_);
  gemm_wmma_bf16<128, false, false><<<dim3(1, M_ / 128), 256, 0, stream>>>(
      abf, wvT, bv, vb, nullptr, C_);

  // edge attention: scores -> segment max -> exp/den -> weighted scatter
  edge_scores_kernel<<<nblk(TEWH, 256), 256, 0, stream>>>(qb, kb, srcp, dstp, sall);
  segmax_kernel<<<nblk(TEH, 256), 256, 0, stream>>>(sall, dstp, mbuf);
  expden_kernel<<<nblk(TEWH, 256), 256, 0, stream>>>(sall, dstp, mbuf, den);
  attn_out_kernel<<<nblk(TEWH, 256), 256, 0, stream>>>(sall, den, spikes, vb, srcp, dstp, aout);

  // msg = out @ Wo + bo
  cvt_f2b_kernel<<<nblk(MC, 256), 256, 0, stream>>>(aout, abf, MC);
  gemm_wmma_bf16<128, false, false><<<dim3(1, M_ / 128), 256, 0, stream>>>(
      abf, woT, bo, msg, nullptr, C_);

  // LIF input current + scan
  imsg_kernel<<<M_ / 8, 256, 0, stream>>>(msg, Wmsg, bmsg, Ibuf, M_);
  lif_kernel<<<nblk(N_, 256), 256, 0, stream>>>(Ibuf, spk_out);

  // FFN: gelu(msg@W1+b1)@W2+b2  (fc1 stores bf16 directly for fc2)
  cvt_f2b_kernel<<<nblk(MC, 256), 256, 0, stream>>>(msg, abf, MC);
  gemm_wmma_bf16<128, true, true><<<dim3(C4_ / 128, M_ / 128), 256, 0, stream>>>(
      abf, wfc1T, bfc1, nullptr, bbf, C4_);
  gemm_wmma_bf16<512, false, false><<<dim3(1, M_ / 128), 256, 0, stream>>>(
      bbf, wfc2T, bfc2, ffn, nullptr, C_);

  // feat = LN(x + ffn)
  add_ln_kernel<<<M_ / 8, 256, 0, stream>>>(x, ffn, gf, bff, feat);
}
